// CompressiveMemory_14104672600876
// MI455X (gfx1250) — compile-verified, matrix operand order checked by reading, not run on hardware
//
#include <hip/hip_runtime.h>
#include <hip/hip_bf16.h>
#include <stdint.h>

// --- ext-vector types (WMMA operands + wide memory ops) ---
typedef __attribute__((ext_vector_type(16))) _Float16 v16h;
typedef __attribute__((ext_vector_type(8)))  _Float16 v8h;
typedef __attribute__((ext_vector_type(4)))  _Float16 v4h;
typedef __attribute__((ext_vector_type(8)))  float    v8f;
typedef __attribute__((ext_vector_type(4)))  float    v4f;

constexpr int kR   = 8;     // fine-memory rows
constexpr int kK   = 8;     // compressed-memory rows
constexpr int kD   = 256;   // feature dim
constexpr int kMT  = 16;    // samples per block (one WMMA M-tile)
constexpr int kKT  = 512;   // GEMM K = 2*D
constexpr int kPit = 520;   // LDS A-row pitch in halves (padding spreads banks)

// ---------------------------------------------------------------------------
// Kernel 0: pack W -> f16 WMMA B-fragments in d_ws.
// Logical Wt[k][n]: k<256 -> conv_w[n, k, 0] ; k>=256 -> conv_w[n, k-256, 1]
// Fragment f = ks*16 + ntile (ks: K-step of 32, ntile: 16 output cols).
// Per lane, 16 halves in the ISA B-layout for v_wmma_f32_16x16x32_f16:
//   lane<16 : n = ntile*16+lane,    K = ks*32 + {0..7, 16..23}
//   lane>=16: n = ntile*16+lane-16, K = ks*32 + {8..15, 24..31}
// ---------------------------------------------------------------------------
__global__ __launch_bounds__(256) void pack_w_frags(const float* __restrict__ conv_w,
                                                    _Float16* __restrict__ wfrag) {
  int gid   = blockIdx.x * 256 + threadIdx.x;  // 8192 total
  int lane  = gid & 31;
  int f     = gid >> 5;                        // 0..255
  int ntile = f & 15;
  int ks    = f >> 4;                          // 0..15
  int n     = ntile * 16 + (lane & 15);
  int khalf = lane >> 4;
  _Float16* dst = wfrag + (size_t)gid * 16;
#pragma unroll
  for (int h = 0; h < 16; ++h) {
    int K = ks * 32 + ((h < 8) ? (khalf * 8 + h) : (16 + khalf * 8 + (h - 8)));
    float w = (K < kD) ? conv_w[((size_t)n * kD + K) * 2 + 0]
                       : conv_w[((size_t)n * kD + (K - kD)) * 2 + 1];
    dst[h] = (_Float16)w;
  }
}

// ---------------------------------------------------------------------------
// Kernel 1: per block of 16 samples:
//   - compute insert/shift flags, write fm_count_new
//   - if any sample shifts: stage A=[cm_last|fm0] in LDS (f16), batched WMMA
//     GEMM -> c_new tile in LDS (f32, +bias)
//   - streamed (non-temporal) b128 copy/permute of fm and cm into outputs
// ---------------------------------------------------------------------------
__global__ __launch_bounds__(256) void cm_update(
    const float* __restrict__ fm, const float* __restrict__ cm,
    const int* __restrict__ fm_count, const float* __restrict__ seg,
    const unsigned char* __restrict__ valid, const float* __restrict__ conv_b,
    const _Float16* __restrict__ wfrag,
    float* __restrict__ fm_out, float* __restrict__ cm_out,
    float* __restrict__ cnt_out) {
  __shared__ _Float16 lA[kMT][kPit];  // A tile, f16  (~16.6 KB)
  __shared__ float    lC[kMT][kD];    // c_new tile, f32 (16 KB)
  __shared__ int s_shift[kMT], s_insert[kMT], s_cnt[kMT];

  const int tid = threadIdx.x;
  const int b0  = blockIdx.x * kMT;

  if (tid < kMT) {
    int b   = b0 + tid;
    int c   = fm_count[b];
    int v   = (valid[b] != 0);
    int ins = v && (c < kR);
    s_cnt[tid]    = c;
    s_insert[tid] = ins;
    s_shift[tid]  = v && (c >= kR);
    cnt_out[b]    = (float)(c + ins);  // d_out is a float buffer
  }
  __syncthreads();

  int any = 0;
#pragma unroll
  for (int m = 0; m < kMT; ++m) any |= s_shift[m];  // block-uniform

  if (any) {
    // ---- stage A tile: 16 rows x 512 (cm_last | fm_row0), f32 -> f16 ----
    for (int i = tid; i < kMT * (kKT / 4); i += 256) {
      int m  = i >> 7;
      int c4 = i & 127;
      size_t b = (size_t)(b0 + m);
      v4f val;
      if (c4 < kD / 4) val = ((const v4f*)(cm + (b * kK + (kK - 1)) * kD))[c4];
      else             val = ((const v4f*)(fm + b * kR * kD))[c4 - kD / 4];
      v4h h;
      h[0] = (_Float16)val[0]; h[1] = (_Float16)val[1];
      h[2] = (_Float16)val[2]; h[3] = (_Float16)val[3];
      *(v4h*)&lA[m][c4 * 4] = h;
    }
    __syncthreads();

    // ---- batched GEMM: each wave owns 2 N-tiles of 16 outputs ----
    const int wave  = tid >> 5;
    const int lane  = tid & 31;
    const int mr    = lane & 15;
    const int khalf = lane >> 4;
#pragma unroll
    for (int t = 0; t < 2; ++t) {
      const int ntile = wave * 2 + t;
      v8f acc = {};
#pragma unroll
      for (int ks = 0; ks < 16; ++ks) {
        const int base = ks * 32 + khalf * 8;
        v8h alo = *(const v8h*)&lA[mr][base];        // ds_load_b128
        v8h ahi = *(const v8h*)&lA[mr][base + 16];   // ds_load_b128
        v16h a;
#pragma unroll
        for (int q = 0; q < 8; ++q) { a[q] = alo[q]; a[q + 8] = ahi[q]; }
        const v8h* bp = (const v8h*)wfrag +
                        (((size_t)(ks * 16 + ntile) * 32) + lane) * 2;
        v8h blo = bp[0];                             // global b128, coalesced
        v8h bhi = bp[1];
        v16h bb;
#pragma unroll
        for (int q = 0; q < 8; ++q) { bb[q] = blo[q]; bb[q + 8] = bhi[q]; }
        acc = __builtin_amdgcn_wmma_f32_16x16x32_f16(
            /*neg_a=*/false, a, /*neg_b=*/false, bb,
            /*c_mod=*/(short)0, acc, /*reuse_a=*/false, /*reuse_b=*/false);
      }
      // epilogue: +bias, scatter into lC per ISA C/D layout
      const int n     = ntile * 16 + mr;
      const float bias = conv_b[n];
      const int mbase = khalf * 8;
#pragma unroll
      for (int j = 0; j < 8; ++j) lC[mbase + j][n] = acc[j] + bias;
    }
  }
  __syncthreads();

  // ---- fm_new: streamed permute/copy (b128, non-temporal) ----
  for (int i = tid; i < kMT * kR * (kD / 4); i += 256) {
    int c  = i & 63;
    int rm = i >> 6;
    int r  = rm & 7;
    int m  = rm >> 3;
    size_t b = (size_t)(b0 + m);
    const v4f* src;
    if (s_shift[m])
      src = (r < kR - 1) ? (const v4f*)(fm + (b * kR + r + 1) * kD)
                         : (const v4f*)(seg + b * kD);
    else if (s_insert[m] && r == s_cnt[m])
      src = (const v4f*)(seg + b * kD);
    else
      src = (const v4f*)(fm + (b * kR + r) * kD);
    v4f v = __builtin_nontemporal_load(src + c);
    __builtin_nontemporal_store(v, (v4f*)(fm_out + (b * kR + r) * kD) + c);
  }

  // ---- cm_new: streamed copy; shifted tail row comes from lC ----
  for (int i = tid; i < kMT * kK * (kD / 4); i += 256) {
    int c  = i & 63;
    int rm = i >> 6;
    int r  = rm & 7;
    int m  = rm >> 3;
    size_t b = (size_t)(b0 + m);
    v4f v;
    if (s_shift[m]) {
      if (r < kK - 1)
        v = __builtin_nontemporal_load((const v4f*)(cm + (b * kK + r + 1) * kD) + c);
      else
        v = *((const v4f*)&lC[m][0] + c);
    } else {
      v = __builtin_nontemporal_load((const v4f*)(cm + (b * kK + r) * kD) + c);
    }
    __builtin_nontemporal_store(v, (v4f*)(cm_out + (b * kK + r) * kD) + c);
  }
}

// ---------------------------------------------------------------------------
// Host launcher. Inputs (setup_inputs order):
//  0 fine_memory (B,8,256) f32   1 compressed_memory (B,8,256) f32
//  2 fm_count (B,) i32           3 new_segment (B,256) f32
//  4 valid_mask (B,) bool(1B)    5 conv_w (256,256,2) f32   6 conv_b (256,) f32
// d_out: [fm_new | cm_new | fm_count_new] flat, float.
// ---------------------------------------------------------------------------
extern "C" void kernel_launch(void* const* d_in, const int* in_sizes, int n_in,
                              void* d_out, int out_size, void* d_ws, size_t ws_size,
                              hipStream_t stream) {
  const float*         fm       = (const float*)d_in[0];
  const float*         cm       = (const float*)d_in[1];
  const int*           fm_count = (const int*)d_in[2];
  const float*         seg      = (const float*)d_in[3];
  const unsigned char* valid    = (const unsigned char*)d_in[4];
  const float*         conv_w   = (const float*)d_in[5];
  const float*         conv_b   = (const float*)d_in[6];

  const int B = in_sizes[2];  // 65536

  float* fm_out  = (float*)d_out;
  float* cm_out  = fm_out + (size_t)B * kR * kD;
  float* cnt_out = cm_out + (size_t)B * kK * kD;

  _Float16* wfrag = (_Float16*)d_ws;  // 256 KB of packed B-fragments

  pack_w_frags<<<32, 256, 0, stream>>>(conv_w, wfrag);
  cm_update<<<B / kMT, 256, 0, stream>>>(fm, cm, fm_count, seg, valid, conv_b,
                                         wfrag, fm_out, cm_out, cnt_out);
}
